// GraphAttentionEmbedding_65627100283652
// MI455X (gfx1250) — matrix operand order repaired and checked by reading, not compiled, not run on hardware
//
#include <hip/hip_runtime.h>

typedef __attribute__((ext_vector_type(16))) __bf16 v16bf;
typedef __attribute__((ext_vector_type(8)))  float  v8f;

#define TGN_B    4096
#define TGN_K    20
#define TGN_D    256
#define TGN_FE   256
#define TGN_T    100
#define TGN_E    356
#define TGN_EP   368    // E padded to 16
#define TGN_KDIM 612
#define TGN_HD   178
#define SRC_PER_WG 4
#define NRR       80    // SRC_PER_WG * TGN_K

// workspace layout (dword offsets) — weights stored fragment-major:
// block(ks,nt) = 32 lanes x 8 dwords contiguous (1KB), ks-major over nt.
enum : unsigned {
  OFF_PROJ = 0,            // 16 ks x 16 nt
  OFF_WQ   = 65536,        // 2 x (12 ks x 23 nt)
  OFF_WK   = 206848,       // 2 x (20 ks x 23 nt)
  OFF_WV   = 442368,
  OFF_WO   = 677888,       // 2 x (12 ks x 23 nt)
  OFF_FC1  = 819200,       // 2 x (20 ks x 16 nt)
  OFF_FC2  = 983040,       // 2 x ( 8 ks x 16 nt)
  OFF_CUR  = 1048576,      // 4096x256 f32
  OFF_TE   = 2097152       // 4096x100 f32
};

__device__ __forceinline__ unsigned packbf2(float lo, float hi) {
  unsigned a = __float_as_uint(lo), b = __float_as_uint(hi);
  unsigned rl = (a + 0x7FFFu + ((a >> 16) & 1u)) >> 16;  // RNE fp32->bf16
  unsigned rh = (b + 0x7FFFu + ((b >> 16) & 1u)) >> 16;
  return rl | (rh << 16);
}

__device__ __forceinline__ float bfbits(unsigned dw, int hi) {
  return __uint_as_float(hi ? (dw & 0xFFFF0000u) : (dw << 16));
}

union FragU { unsigned u[8]; v16bf v; };

// A fragment (16x32 bf16) from LDS, pair-packed rows [m][k/2] dwords.
// lane<16: rows m, k=0..7 & 16..23 ; lane>=16: k=8..15 & 24..31  (ISA 7.12.2)
__device__ __forceinline__ v16bf load_a_frag(const unsigned* lds, int rowStride,
                                             int m0, int k2base, int lane) {
  FragU f;
  const unsigned* row = lds + (m0 + (lane & 15)) * rowStride + k2base + ((lane >> 4) << 2);
#pragma unroll
  for (int i = 0; i < 8; i++) f.u[i] = row[((i >> 2) << 3) + (i & 3)];
  return f.v;
}

// B fragment from fragment-major weights: contiguous 32B per lane -> b128 x2.
__device__ __forceinline__ v16bf load_b_frag(const unsigned* __restrict__ w,
                                             int ntiles, int ks, int nt, int lane) {
  FragU f;
  const unsigned* p = w + ((unsigned)(ks * ntiles + nt) * 32 + lane) * 8;
#pragma unroll
  for (int i = 0; i < 8; i++) f.u[i] = p[i];
  return f.v;
}

__device__ __forceinline__ void gemm_tile(const unsigned* aLds, int aStride, int m0,
                                          const unsigned* __restrict__ bW, int ntiles,
                                          int nt, int ksteps, int lane, v8f& acc) {
  for (int ks = 0; ks < ksteps; ks++) {
    if (ks + 1 < ksteps)  // cover L2 latency on the weight stream
      __builtin_prefetch(bW + ((unsigned)((ks + 1) * ntiles + nt) * 32 + lane) * 8, 0, 3);
    v16bf a = load_a_frag(aLds, aStride, m0, ks * 16, lane);
    v16bf b = load_b_frag(bW, ntiles, ks, nt, lane);
    acc = __builtin_amdgcn_wmma_f32_16x16x32_bf16(false, a, false, b, (short)0, acc,
                                                  false, false);
  }
}

#define ZERO8 {0.f,0.f,0.f,0.f,0.f,0.f,0.f,0.f}

// ---------------------------------------------------------------------------
// Kernel 1: pack all weights to bf16 fragment-major B layout (zero padded)
// ---------------------------------------------------------------------------
__global__ __launch_bounds__(256) void tgn_convert_weights(
    const float* __restrict__ projW, const float* __restrict__ Wq,
    const float* __restrict__ Wk, const float* __restrict__ Wv,
    const float* __restrict__ Wo, const float* __restrict__ fc1W,
    const float* __restrict__ fc2W, unsigned* __restrict__ wsu) {
  int job = blockIdx.y;
  const float* W; unsigned* out;
  int ksteps, ntiles, realK, realN, ld, tr;
  if (job == 0)      { W = projW;                     out = wsu + OFF_PROJ;             ksteps = 16; ntiles = 16; realK = 512; realN = 256; ld = 256; tr = 0; }
  else if (job <= 2) { int l = job - 1;  W = Wq   + l * TGN_E * TGN_E;    out = wsu + OFF_WQ  + l * 70656;  ksteps = 12; ntiles = 23; realK = 356; realN = 356; ld = 356; tr = 1; }
  else if (job <= 4) { int l = job - 3;  W = Wk   + l * TGN_E * TGN_KDIM; out = wsu + OFF_WK  + l * 117760; ksteps = 20; ntiles = 23; realK = 612; realN = 356; ld = 612; tr = 1; }
  else if (job <= 6) { int l = job - 5;  W = Wv   + l * TGN_E * TGN_KDIM; out = wsu + OFF_WV  + l * 117760; ksteps = 20; ntiles = 23; realK = 612; realN = 356; ld = 612; tr = 1; }
  else if (job <= 8) { int l = job - 7;  W = Wo   + l * TGN_E * TGN_E;    out = wsu + OFF_WO  + l * 70656;  ksteps = 12; ntiles = 23; realK = 356; realN = 356; ld = 356; tr = 1; }
  else if (job <= 10){ int l = job - 9;  W = fc1W + l * TGN_D * TGN_KDIM; out = wsu + OFF_FC1 + l * 81920;  ksteps = 20; ntiles = 16; realK = 612; realN = 256; ld = 612; tr = 1; }
  else               { int l = job - 11; W = fc2W + l * TGN_D * TGN_D;    out = wsu + OFF_FC2 + l * 32768;  ksteps = 8;  ntiles = 16; realK = 256; realN = 256; ld = 256; tr = 1; }
  int total = ksteps * ntiles * 256;
  for (int i = blockIdx.x * blockDim.x + threadIdx.x; i < total;
       i += gridDim.x * blockDim.x) {
    int i8 = i & 7, lane = (i >> 3) & 31;
    int rest = i >> 8;
    int nt = rest % ntiles, ks = rest / ntiles;
    int k0 = ks * 32 + ((lane >> 4) << 3) + ((i8 >> 2) << 4) + 2 * (i8 & 3);   // B-frag element map
    int n  = nt * 16 + (lane & 15);
    float lo = 0.f, hi = 0.f;
    if (n < realN) {
      if (k0 < realK)     lo = tr ? W[n * ld + k0]     : W[k0 * ld + n];
      if (k0 + 1 < realK) hi = tr ? W[n * ld + k0 + 1] : W[(k0 + 1) * ld + n];
    }
    out[i] = packbf2(lo, hi);
  }
}

// ---------------------------------------------------------------------------
// Kernel 2: source init: cur0 = [nf||mem][src] @ projW + b ; src_te = cos(t*w+b)
// ---------------------------------------------------------------------------
__global__ __launch_bounds__(256) void tgn_src_init(
    const float* __restrict__ nfeat, const float* __restrict__ memt,
    const int* __restrict__ srcn, const float* __restrict__ ts,
    const float* __restrict__ time_w, const float* __restrict__ time_b,
    const float* __restrict__ proj_b, const unsigned* __restrict__ projB,
    float* __restrict__ cur, float* __restrict__ te) {
  __shared__ unsigned sh_a[32 * 256];
  const int tid = threadIdx.x, wave = tid >> 5, lane = tid & 31;
  const int m0 = blockIdx.x * 32;
  for (int idx = tid; idx < 32 * TGN_T; idx += 256) {
    int r = idx / TGN_T, t = idx % TGN_T;
    te[(m0 + r) * TGN_T + t] = cosf(ts[m0 + r] * time_w[t] + time_b[t]);
  }
  for (int idx = tid; idx < 32 * 256; idx += 256) {
    int r = idx >> 8, cdw = idx & 255;
    long s = srcn[m0 + r];
    float lo, hi;
    if (cdw < 128) { lo = nfeat[s * 256 + 2 * cdw]; hi = nfeat[s * 256 + 2 * cdw + 1]; }
    else { int q = cdw - 128; lo = memt[s * 256 + 2 * q]; hi = memt[s * 256 + 2 * q + 1]; }
    sh_a[idx] = packbf2(lo, hi);
  }
  __syncthreads();
  for (int t = wave; t < 32; t += 8) {   // 2 M-tiles x 16 N-tiles
    int mt = t >> 4, nt = t & 15;
    v8f acc = ZERO8;
    gemm_tile(sh_a, 256, mt * 16, projB, 16, nt, 16, lane, acc);
    int n = nt * 16 + (lane & 15);
    float bias = proj_b[n];
#pragma unroll
    for (int i = 0; i < 8; i++) {
      int m = mt * 16 + ((lane >> 4) << 3) + i;
      cur[(m0 + m) * 256 + n] = acc[i] + bias;
    }
  }
}

// ---------------------------------------------------------------------------
// Kernel 3: one fused attention layer. WG = 4 sources (80 neighbor rows).
// ---------------------------------------------------------------------------
__global__ __launch_bounds__(256) void tgn_layer(
    int layer, const float* __restrict__ cur_in, float* __restrict__ cur_out,
    const float* __restrict__ nfeat, const float* __restrict__ memt,
    const float* __restrict__ efeat, const int* __restrict__ nbrs,
    const int* __restrict__ eidx, const float* __restrict__ etimes,
    const float* __restrict__ time_w, const float* __restrict__ time_b,
    const float* __restrict__ proj_b, const float* __restrict__ bq,
    const float* __restrict__ bk, const float* __restrict__ bv,
    const float* __restrict__ bo, const float* __restrict__ fc1b,
    const float* __restrict__ fc2b, const unsigned* __restrict__ wsu,
    const float* __restrict__ te) {
  __shared__ unsigned sh_kin[NRR * 320];   // k_in, 640 bf16 cols pair-packed
  __shared__ unsigned sh_ktv[NRR * 184];   // Kt then V, 368 bf16 cols
  __shared__ unsigned sh_a1[16 * 256];     // gather staging (512 bf16 cols)
  __shared__ unsigned sh_qbf[16 * 192];    // q_in bf16, rows 4..15 zero
  __shared__ float    sh_qf[4 * TGN_EP];   // Q fp32
  __shared__ float    sh_sc[4 * 2 * TGN_K];// scores -> attn weights
  __shared__ int      sh_inv[4];
  __shared__ int      sh_nid[NRR];
  // overlays into dead k_in region (k_in unused after V GEMM)
  unsigned* ctxA  = sh_kin;                 // 16 x 192 dw
  unsigned* fc1in = sh_kin + 16 * 192;      // 16 x 320 dw
  unsigned* hbf   = fc1in + 16 * 320;       // 16 x 128 dw

  const unsigned* projB = wsu + OFF_PROJ;
  const unsigned* wqB = wsu + OFF_WQ + layer * 70656;
  const unsigned* wkB = wsu + OFF_WK + layer * 117760;
  const unsigned* wvB = wsu + OFF_WV + layer * 117760;
  const unsigned* woB = wsu + OFF_WO + layer * 70656;
  const unsigned* f1B = wsu + OFF_FC1 + layer * 81920;
  const unsigned* f2B = wsu + OFF_FC2 + layer * 32768;

  const int tid = threadIdx.x, wave = tid >> 5, lane = tid & 31;
  const int s0 = blockIdx.x * SRC_PER_WG;

  // -------- phase 0a: neighbor ids, q_in, k_in[edge||time] ----------------
  for (int j = tid; j < NRR; j += 256) {
    int src = s0 + j / TGN_K, kk = j % TGN_K;
    sh_nid[j] = nbrs[((long)layer * TGN_B + src) * TGN_K + kk];
  }
  for (int idx = tid; idx < 16 * 192; idx += 256) {
    int r = idx / 192, c = idx % 192;
    unsigned v = 0;
    if (r < 4) {
      int src = s0 + r;
      if (c < 128) v = packbf2(cur_in[src * TGN_D + 2 * c], cur_in[src * TGN_D + 2 * c + 1]);
      else if (c < 178) {
        int t0 = (c - 128) * 2;
        v = packbf2(te[src * TGN_T + t0], te[src * TGN_T + t0 + 1]);
      }
    }
    sh_qbf[idx] = v;
  }
  for (int idx = tid; idx < NRR * 192; idx += 256) {
    int j = idx / 192, c = idx % 192;
    int src = s0 + j / TGN_K, kk = j % TGN_K;
    long base = ((long)layer * TGN_B + src) * TGN_K + kk;
    unsigned v = 0;
    if (c < 128) {
      long e = eidx[base];
      v = packbf2(efeat[e * TGN_FE + 2 * c], efeat[e * TGN_FE + 2 * c + 1]);
    } else if (c < 178) {
      float t = etimes[base];
      int t0 = (c - 128) * 2;
      v = packbf2(cosf(t * time_w[t0] + time_b[t0]),
                  cosf(t * time_w[t0 + 1] + time_b[t0 + 1]));
    }
    sh_kin[j * 320 + 128 + c] = v;
  }
  __syncthreads();
  if (tid < 4) {
    int allz = 1;
    for (int kk = 0; kk < TGN_K; kk++) allz &= (sh_nid[tid * TGN_K + kk] == 0);
    sh_inv[tid] = allz;
  }
  // -------- phase 0b: Q = q_in @ Wq^T + bq (fp32 into LDS) ----------------
  for (int t = wave; t < 23; t += 8) {
    v8f acc = ZERO8;
    gemm_tile(sh_qbf, 192, 0, wqB, 23, t, 12, lane, acc);
    int n = t * 16 + (lane & 15);
    float bias = (n < TGN_E) ? bq[layer * TGN_E + n] : 0.f;
#pragma unroll
    for (int i = 0; i < 8; i++) {
      int m = ((lane >> 4) << 3) + i;
      if (m < 4) sh_qf[m * TGN_EP + n] = acc[i] + bias;
    }
  }
  __syncthreads();
  // -------- phase 1: neighbor proj GEMM into k_in[:,0:256] ----------------
  for (int c = 0; c < 5; c++) {
    for (int idx = tid; idx < 16 * 256; idx += 256) {
      int r = idx >> 8, cdw = idx & 255;
      long node = sh_nid[c * 16 + r];
      float lo, hi;
      if (cdw < 128) { lo = nfeat[node * 256 + 2 * cdw]; hi = nfeat[node * 256 + 2 * cdw + 1]; }
      else { int q = cdw - 128; lo = memt[node * 256 + 2 * q]; hi = memt[node * 256 + 2 * q + 1]; }
      sh_a1[idx] = packbf2(lo, hi);
    }
    __syncthreads();
    for (int t = wave; t < 16; t += 8) {
      v8f acc = ZERO8;
      gemm_tile(sh_a1, 256, 0, projB, 16, t, 16, lane, acc);
      int n = t * 16 + (lane & 15);
      float bias = proj_b[n];
#pragma unroll
      for (int i = 0; i < 8; i++) {
        float v = acc[i] + bias;
        float v2 = __shfl_xor(v, 1);
        if ((lane & 1) == 0) {
          int m = ((lane >> 4) << 3) + i;
          sh_kin[(c * 16 + m) * 320 + (n >> 1)] = packbf2(v, v2);
        }
      }
    }
    __syncthreads();
  }
  // -------- phase 2: Kt = k_in @ Wk^T + bk (bf16 into LDS) ----------------
  for (int t = wave; t < 5 * 23; t += 8) {
    int mt = t / 23, nt = t % 23;
    v8f acc = ZERO8;
    gemm_tile(sh_kin, 320, mt * 16, wkB, 23, nt, 20, lane, acc);
    int n = nt * 16 + (lane & 15);
    float bias = (n < TGN_E) ? bk[layer * TGN_E + n] : 0.f;
#pragma unroll
    for (int i = 0; i < 8; i++) {
      float v = acc[i] + bias;
      float v2 = __shfl_xor(v, 1);
      if ((lane & 1) == 0) {
        int m = mt * 16 + ((lane >> 4) << 3) + i;
        sh_ktv[m * 184 + (n >> 1)] = packbf2(v, v2);
      }
    }
  }
  __syncthreads();
  // -------- phase 3: scores + mask + softmax ------------------------------
  if (tid < 160) {
    int src = tid / 40, rem = tid % 40, h = rem / 20, kk = rem % 20;
    int row = src * TGN_K + kk;
    float dot = 0.f;
    for (int d = 0; d < TGN_HD; d++) {
      int n = h * TGN_HD + d;
      dot += sh_qf[src * TGN_EP + n] * bfbits(sh_ktv[row * 184 + (n >> 1)], n & 1);
    }
    dot *= 7.4953169e-2f;  // 1/sqrt(178)
    int masked = (sh_nid[row] == 0) && !(sh_inv[src] && kk == 0);
    sh_sc[(src * 2 + h) * TGN_K + kk] = masked ? -1e9f : dot;
  }
  __syncthreads();
  if (tid < 8) {
    float* s = &sh_sc[tid * TGN_K];
    float mx = s[0];
    for (int k = 1; k < TGN_K; k++) mx = fmaxf(mx, s[k]);
    float sum = 0.f;
    for (int k = 0; k < TGN_K; k++) { float e = __expf(s[k] - mx); s[k] = e; sum += e; }
    float inv = 1.f / sum;
    for (int k = 0; k < TGN_K; k++) s[k] *= inv;
  }
  __syncthreads();
  // -------- phase 4: V = k_in @ Wv^T + bv (overwrites Kt region) ----------
  for (int t = wave; t < 5 * 23; t += 8) {
    int mt = t / 23, nt = t % 23;
    v8f acc = ZERO8;
    gemm_tile(sh_kin, 320, mt * 16, wvB, 23, nt, 20, lane, acc);
    int n = nt * 16 + (lane & 15);
    float bias = (n < TGN_E) ? bv[layer * TGN_E + n] : 0.f;
#pragma unroll
    for (int i = 0; i < 8; i++) {
      float v = acc[i] + bias;
      float v2 = __shfl_xor(v, 1);
      if ((lane & 1) == 0) {
        int m = mt * 16 + ((lane >> 4) << 3) + i;
        sh_ktv[m * 184 + (n >> 1)] = packbf2(v, v2);
      }
    }
  }
  __syncthreads();
  // -------- phase 5: ctx = attn @ V (bf16 A for Wo) + fc1in prefill -------
  for (int idx = tid; idx < 16 * 192; idx += 256) {
    int r = idx / 192, cdw = idx % 192;
    unsigned v = 0;
    if (r < 4 && cdw < 178) {
      int n0 = cdw * 2;
      int h = (n0 >= TGN_HD) ? 1 : 0;
      const float* att = &sh_sc[(r * 2 + h) * TGN_K];
      float a0 = 0.f, a1 = 0.f;
      for (int kk = 0; kk < TGN_K; kk++) {
        unsigned dw = sh_ktv[(r * TGN_K + kk) * 184 + cdw];
        float w = att[kk];
        a0 += w * bfbits(dw, 0);
        a1 += w * bfbits(dw, 1);
      }
      v = packbf2(a0, a1);
    }
    ctxA[idx] = v;
  }
  for (int idx = tid; idx < 16 * 320; idx += 256) {
    int r = idx / 320, cdw = idx % 320;
    if (r < 4 && cdw < 178) continue;  // written by Wo phase
    unsigned v = 0;
    if (r < 4 && cdw >= 178 && cdw < 306) {
      int j2 = (cdw - 178) * 2, src = s0 + r;
      v = packbf2(cur_in[src * TGN_D + j2], cur_in[src * TGN_D + j2 + 1]);
    }
    fc1in[idx] = v;
  }
  __syncthreads();
  // -------- phase 6: attn_out = ctx @ Wo^T + bo (zero if invalid) ---------
  for (int t = wave; t < 23; t += 8) {
    v8f acc = ZERO8;
    gemm_tile(ctxA, 192, 0, woB, 23, t, 12, lane, acc);
    int n = t * 16 + (lane & 15);
    float bias = (n < TGN_E) ? bo[layer * TGN_E + n] : 0.f;
#pragma unroll
    for (int i = 0; i < 8; i++) {
      int m = ((lane >> 4) << 3) + i;
      float v = acc[i] + bias;
      if (m < 4 && sh_inv[m]) v = 0.f;
      float v2 = __shfl_xor(v, 1);
      if ((lane & 1) == 0 && m < 4 && n < TGN_E)
        fc1in[m * 320 + (n >> 1)] = packbf2(v, v2);
    }
  }
  __syncthreads();
  // -------- phase 7: h = relu([attn_out||cur] @ fc1^T + b) ----------------
  for (int t = wave; t < 16; t += 8) {
    v8f acc = ZERO8;
    gemm_tile(fc1in, 320, 0, f1B, 16, t, 20, lane, acc);
    int n = t * 16 + (lane & 15);
    float bias = fc1b[layer * TGN_D + n];
#pragma unroll
    for (int i = 0; i < 8; i++) {
      float v = fmaxf(acc[i] + bias, 0.f);
      float v2 = __shfl_xor(v, 1);
      if ((lane & 1) == 0) {
        int m = ((lane >> 4) << 3) + i;
        hbf[m * 128 + (n >> 1)] = packbf2(v, v2);
      }
    }
  }
  __syncthreads();
  // -------- phase 8: cur' = h @ fc2^T + b ---------------------------------
  for (int t = wave; t < 16; t += 8) {
    v8f acc = ZERO8;
    gemm_tile(hbf, 128, 0, f2B, 16, t, 8, lane, acc);
    int n = t * 16 + (lane & 15);
    float bias = fc2b[layer * TGN_D + n];
#pragma unroll
    for (int i = 0; i < 8; i++) {
      int m = ((lane >> 4) << 3) + i;
      if (m < 4) cur_out[(s0 + m) * TGN_D + n] = acc[i] + bias;
    }
  }
}

// ---------------------------------------------------------------------------
extern "C" void kernel_launch(void* const* d_in, const int* in_sizes, int n_in,
                              void* d_out, int out_size, void* d_ws, size_t ws_size,
                              hipStream_t stream) {
  const float* nfeat  = (const float*)d_in[0];
  const float* memt   = (const float*)d_in[1];
  const float* efeat  = (const float*)d_in[2];
  const int*   srcn   = (const int*)d_in[3];
  const float* ts     = (const float*)d_in[4];
  const int*   nbrs   = (const int*)d_in[5];
  const int*   eidx   = (const int*)d_in[6];
  const float* etimes = (const float*)d_in[7];
  const float* time_w = (const float*)d_in[8];
  const float* time_b = (const float*)d_in[9];
  const float* projW  = (const float*)d_in[10];
  const float* proj_b = (const float*)d_in[11];
  const float* Wq     = (const float*)d_in[12];
  const float* bq     = (const float*)d_in[13];
  const float* Wk     = (const float*)d_in[14];
  const float* bk     = (const float*)d_in[15];
  const float* Wv     = (const float*)d_in[16];
  const float* bv     = (const float*)d_in[17];
  const float* Wo     = (const float*)d_in[18];
  const float* bo     = (const float*)d_in[19];
  const float* fc1W   = (const float*)d_in[20];
  const float* fc1b   = (const float*)d_in[21];
  const float* fc2W   = (const float*)d_in[22];
  const float* fc2b   = (const float*)d_in[23];

  unsigned* wsu = (unsigned*)d_ws;
  float* cur = (float*)(wsu + OFF_CUR);
  float* te  = (float*)(wsu + OFF_TE);
  float* out = (float*)d_out;

  tgn_convert_weights<<<dim3(96, 13), 256, 0, stream>>>(projW, Wq, Wk, Wv, Wo,
                                                        fc1W, fc2W, wsu);
  tgn_src_init<<<TGN_B / 32, 256, 0, stream>>>(nfeat, memt, srcn, ts, time_w,
                                               time_b, proj_b, wsu + OFF_PROJ,
                                               cur, te);
  tgn_layer<<<TGN_B / SRC_PER_WG, 256, 0, stream>>>(
      0, cur, cur, nfeat, memt, efeat, nbrs, eidx, etimes, time_w, time_b,
      proj_b, bq, bk, bv, bo, fc1b, fc2b, wsu, te);
  tgn_layer<<<TGN_B / SRC_PER_WG, 256, 0, stream>>>(
      1, cur, out, nfeat, memt, efeat, nbrs, eidx, etimes, time_w, time_b,
      proj_b, bq, bk, bv, bo, fc1b, fc2b, wsu, te);
}
